// Attention_69638599737718
// MI455X (gfx1250) — matrix operand order repaired and checked by reading, not compiled, not run on hardware
//
#include <hip/hip_runtime.h>

// Problem dims (all fixed by the reference)
#define T_DIM  4096
#define NE_DIM 2048
#define DA_DIM 2048
#define NCH    64   // number of scan chunks
#define LCH    64   // chunk length (NCH*LCH == T_DIM)

typedef __attribute__((ext_vector_type(16))) __bf16        v16bf;
typedef __attribute__((ext_vector_type(8)))  float         v8f;
typedef __attribute__((ext_vector_type(4)))  unsigned int  u32x4;

struct Frag32 { u32x4 lo, hi; };   // 32 bytes == one v16bf operand

static __device__ __forceinline__ unsigned short f2bf(float f) {
  union { float f; unsigned u; } cv; cv.f = f;
  unsigned u = cv.u;
  unsigned r = (u + 0x7FFFu + ((u >> 16) & 1u)) >> 16;  // RNE
  return (unsigned short)r;
}

// LDS byte offset of a shared-memory pointer (low 32 bits of the generic VA)
static __device__ __forceinline__ unsigned lds_off(const void* p) {
  return (unsigned)(unsigned long long)p;
}

// CDNA5 async copy: global -> LDS, 16B per lane, tracked by ASYNCcnt
static __device__ __forceinline__ void async_b128(unsigned lds, unsigned long long gaddr) {
  asm volatile("global_load_async_to_lds_b128 %0, %1, off"
               :: "v"(lds), "v"(gaddr) : "memory");
}

// ---------------------------------------------------------------------------
// Transpose + f32 -> bf16 weight conversion: W[2048][2048] -> Wt[c][r] bf16
// ---------------------------------------------------------------------------
__global__ void transpose_to_bf16(const float* __restrict__ W,
                                  unsigned short* __restrict__ Wt) {
  long idx = (long)blockIdx.x * blockDim.x + threadIdx.x;  // < 2048*2048
  int r = (int)(idx >> 11);
  int c = (int)(idx & 2047);
  Wt[(size_t)c * 2048 + r] = f2bf(W[idx]);
}

// ---------------------------------------------------------------------------
// Time-shift mixing, producing bf16 activations for the three projections
// ---------------------------------------------------------------------------
__global__ void mix_bf16(const float* __restrict__ x,
                         const float* __restrict__ mk,
                         const float* __restrict__ mv,
                         const float* __restrict__ mr,
                         unsigned short* __restrict__ xk,
                         unsigned short* __restrict__ xv,
                         unsigned short* __restrict__ xr) {
  long idx = (long)blockIdx.x * blockDim.x + threadIdx.x;  // < T*NE
  int c = (int)(idx & (NE_DIM - 1));
  float xc = x[idx];
  float xp = (idx >= NE_DIM) ? x[idx - NE_DIM] : 0.0f;
  float tk = mk[c], tv = mv[c], tr = mr[c];
  xk[idx] = f2bf(xc * tk + xp * (1.0f - tk));
  xv[idx] = f2bf(xc * tv + xp * (1.0f - tv));
  xr[idx] = f2bf(xc * tr + xp * (1.0f - tr));
}

// ---------------------------------------------------------------------------
// bf16 GEMM with f32 accumulate: C[M][N] = A[M][K] * Bt[N][K]^T
// M=4096, N=2048, K=2048 (all four GEMMs share this shape).
// Block: 256 threads (8 waves), 128x128 tile, K-step 32, double-buffered LDS
// filled by CDNA5 async global->LDS DMA (ASYNCcnt), overlapped with WMMA.
// Wave grid 2(M) x 4(N): each wave -> 64x32 = 4x2 tiles of 16x16.
// ---------------------------------------------------------------------------
__global__ __launch_bounds__(256) void gemm_bf16(
    const unsigned short* __restrict__ A,
    const unsigned short* __restrict__ Bt,
    float* __restrict__ C) {
  constexpr int N = 2048, K = 2048;
  // 40-ushort row stride = 80 bytes: keeps 16B alignment, staggers banks
  __shared__ __align__(16) unsigned short As[2][128][40];
  __shared__ __align__(16) unsigned short Bs[2][128][40];

  const int tid  = threadIdx.x;
  const int m0   = blockIdx.y * 128;
  const int n0   = blockIdx.x * 128;
  const int wave = tid >> 5, lane = tid & 31;
  const int wm   = wave & 1,  wn  = wave >> 1;           // 2 x 4 wave grid
  const int lr   = lane & 15, lh  = lane >> 4;

  // This thread's two 16B chunks per operand tile (static, no exec-mask loop)
  const int row0 = tid >> 2,          seg0 = tid & 3;          // chunk tid
  const int row1 = (tid + 256) >> 2,  seg1 = tid & 3;          // chunk tid+256

  const unsigned short* Arow0 = A  + (size_t)(m0 + row0) * K + seg0 * 8;
  const unsigned short* Arow1 = A  + (size_t)(m0 + row1) * K + seg1 * 8;
  const unsigned short* Brow0 = Bt + (size_t)(n0 + row0) * K + seg0 * 8;
  const unsigned short* Brow1 = Bt + (size_t)(n0 + row1) * K + seg1 * 8;

  v8f acc[4][2];
#pragma unroll
  for (int i = 0; i < 4; ++i)
#pragma unroll
    for (int j = 0; j < 2; ++j) acc[i][j] = {};

  // Issue one stage: 4 async b128 loads per thread (2 A chunks + 2 B chunks)
  auto issue = [&](int buf, int k0) {
    async_b128(lds_off(&As[buf][row0][seg0 * 8]),
               (unsigned long long)(Arow0 + k0));
    async_b128(lds_off(&Bs[buf][row0][seg0 * 8]),
               (unsigned long long)(Brow0 + k0));
    async_b128(lds_off(&As[buf][row1][seg1 * 8]),
               (unsigned long long)(Arow1 + k0));
    async_b128(lds_off(&Bs[buf][row1][seg1 * 8]),
               (unsigned long long)(Brow1 + k0));
  };

  issue(0, 0);  // prologue: stage 0 in flight (ASYNCcnt = 4)

  for (int kb = 0; kb < K / 32; ++kb) {
    const int buf = kb & 1;
    // Prefetch next stage into the other buffer (last iter: harmless re-issue
    // of the final tile into the dead buffer; s_endpgm drains counters).
    const int knext = (kb < K / 32 - 1) ? (kb + 1) * 32 : kb * 32;
    issue(buf ^ 1, knext);

    // Wait for the 4 oldest (this stage); leave next stage's 4 in flight.
    asm volatile("s_wait_asynccnt 0x4" ::: "memory");
    __syncthreads();

    // A fragments: lane holds row m=lr, K pairs per ISA 16-bit A layout:
    //   VGPR0-3 -> K = lh*8 .. lh*8+7 ; VGPR4-7 -> K = 16+lh*8 ..
    v16bf af[4];
#pragma unroll
    for (int mt = 0; mt < 4; ++mt) {
      int row = wm * 64 + mt * 16 + lr;
      Frag32 f;
      f.lo = *(const u32x4*)&As[buf][row][lh * 8];
      f.hi = *(const u32x4*)&As[buf][row][16 + lh * 8];
      af[mt] = __builtin_bit_cast(v16bf, f);
    }
    // B fragments: lane holds col n=lr, K = lh*8 .. lh*8+7 and 16+lh*8 .. 16+lh*8+7
    v16bf bfr[2];
#pragma unroll
    for (int nt = 0; nt < 2; ++nt) {
      int row = wn * 32 + nt * 16 + lr;
      Frag32 f;
      f.lo = *(const u32x4*)&Bs[buf][row][lh * 8];
      f.hi = *(const u32x4*)&Bs[buf][row][16 + lh * 8];
      bfr[nt] = __builtin_bit_cast(v16bf, f);
    }

#pragma unroll
    for (int mt = 0; mt < 4; ++mt)
#pragma unroll
      for (int nt = 0; nt < 2; ++nt)
        acc[mt][nt] = __builtin_amdgcn_wmma_f32_16x16x32_bf16(
            false, af[mt], false, bfr[nt], (short)0, acc[mt][nt], false, false);

    // All waves done reading buf before anyone's next issue overwrites it.
    __syncthreads();
  }

  // Epilogue: C layout per ISA (VGPR r -> M = r + lh*8, lane lr -> N)
#pragma unroll
  for (int mt = 0; mt < 4; ++mt)
#pragma unroll
    for (int nt = 0; nt < 2; ++nt) {
      int m = m0 + wm * 64 + mt * 16 + lh * 8;
      int n = n0 + wn * 32 + nt * 16 + lr;
#pragma unroll
      for (int rr = 0; rr < 8; ++rr)
        C[(size_t)(m + rr) * N + n] = acc[mt][nt][rr];
    }
}

// ---------------------------------------------------------------------------
// Chunked parallel WKV scan. The recurrence S_t = e^w * S_{t-1} + e^{k_t}(v_t,1)
// is affine in S, so a length-L chunk composes as S -> e^{wL} * S + C_chunk.
// State kept log-normalized as (p, q, o) ~ (p*e^o, q*e^o).
// ---------------------------------------------------------------------------

// Pass 1: per (chunk, channel), run the state update from the zero state and
// store the chunk contribution (p,q,o). 64*2048 independent threads.
__global__ __launch_bounds__(256) void wkv_chunk_state(
    const float* __restrict__ k, const float* __restrict__ v,
    const float* __restrict__ time_decay,
    float* __restrict__ cp, float* __restrict__ cq, float* __restrict__ co) {
  int gid   = blockIdx.x * 256 + threadIdx.x;   // < NCH * DA
  int chunk = gid >> 11;                        // / DA_DIM
  int ch    = gid & (DA_DIM - 1);
  float w = -__expf(time_decay[ch]);
  float p = 0.0f, q = 0.0f, o = -1e38f;
  int t0 = chunk * LCH;
  for (int j = 0; j < LCH; ++j) {
    size_t i = (size_t)(t0 + j) * DA_DIM + ch;
    float kt = k[i], vt = v[i];
    float no = fmaxf(w + o, kt);
    float A  = __expf(w + o - no);
    float B  = __expf(kt - no);
    p = A * p + B * vt;
    q = A * q + B;
    o = no;
  }
  cp[gid] = p; cq[gid] = q; co[gid] = o;
}

// Pass 2: per channel, exclusive combine across the 64 chunks (serial depth
// NCH). Overwrites (cp,cq,co) with each chunk's *incoming* state.
__global__ __launch_bounds__(256) void wkv_combine(
    const float* __restrict__ time_decay,
    float* __restrict__ cp, float* __restrict__ cq, float* __restrict__ co) {
  int ch = blockIdx.x * 256 + threadIdx.x;  // < DA
  float wL = -__expf(time_decay[ch]) * (float)LCH;
  float p = 0.0f, q = 0.0f, o = -1e38f;
  for (int c = 0; c < NCH; ++c) {
    int idx = c * DA_DIM + ch;
    float pc = cp[idx], qc = cq[idx], oc = co[idx];
    cp[idx] = p; cq[idx] = q; co[idx] = o;    // exclusive incoming state
    float od = o + wL;                         // decay^L on running state
    float no = fmaxf(od, oc);
    float A  = __expf(od - no);
    float B  = __expf(oc - no);
    p = A * p + B * pc;
    q = A * q + B * qc;
    o = no;
  }
}

// Pass 3: per (chunk, channel), replay the chunk from its incoming state,
// emitting sigmoid(r)*wkv in bf16 for the output projection.
__global__ __launch_bounds__(256) void wkv_chunk_apply(
    const float* __restrict__ k, const float* __restrict__ v,
    const float* __restrict__ r,
    const float* __restrict__ time_decay, const float* __restrict__ time_first,
    const float* __restrict__ cp, const float* __restrict__ cq,
    const float* __restrict__ co,
    unsigned short* __restrict__ srwkv) {
  int gid   = blockIdx.x * 256 + threadIdx.x;   // < NCH * DA
  int chunk = gid >> 11;
  int ch    = gid & (DA_DIM - 1);
  float w = -__expf(time_decay[ch]);
  float u = time_first[ch];
  float p = cp[gid], q = cq[gid], o = co[gid];
  int t0 = chunk * LCH;
  for (int j = 0; j < LCH; ++j) {
    size_t i = (size_t)(t0 + j) * DA_DIM + ch;
    float kt = k[i], vt = v[i], rt = r[i];
    // output at t (bonus u on current token)
    float no = fmaxf(o, u + kt);
    float Aa = __expf(o - no);
    float Bb = __expf(u + kt - no);
    float y  = (Aa * p + Bb * vt) / (Aa * q + Bb);
    // state update
    float no2 = fmaxf(w + o, kt);
    float A2  = __expf(w + o - no2);
    float B2  = __expf(kt - no2);
    p = A2 * p + B2 * vt;
    q = A2 * q + B2;
    o = no2;
    float sr = 1.0f / (1.0f + __expf(-rt));
    srwkv[i] = f2bf(sr * y);
  }
}

// ---------------------------------------------------------------------------
extern "C" void kernel_launch(void* const* d_in, const int* in_sizes, int n_in,
                              void* d_out, int out_size, void* d_ws,
                              size_t ws_size, hipStream_t stream) {
  (void)in_sizes; (void)n_in; (void)out_size; (void)ws_size;
  const float* x          = (const float*)d_in[0];
  const float* time_first = (const float*)d_in[1];
  const float* time_decay = (const float*)d_in[2];
  const float* mk         = (const float*)d_in[3];
  const float* mv         = (const float*)d_in[4];
  const float* mr         = (const float*)d_in[5];
  const float* Wk         = (const float*)d_in[6];
  const float* Wv         = (const float*)d_in[7];
  const float* Wr         = (const float*)d_in[8];
  const float* Wo         = (const float*)d_in[9];
  float* out = (float*)d_out;

  char* ws = (char*)d_ws;
  const size_t MB = 1024 * 1024;
  unsigned short* Wkt   = (unsigned short*)(ws + 0 * MB);    //  8 MB [DA][NE]
  unsigned short* Wvt   = (unsigned short*)(ws + 8 * MB);    //  8 MB
  unsigned short* Wrt   = (unsigned short*)(ws + 16 * MB);   //  8 MB
  unsigned short* Wot   = (unsigned short*)(ws + 24 * MB);   //  8 MB [NE][DA]
  unsigned short* xk    = (unsigned short*)(ws + 32 * MB);   // 16 MB [T][NE]
  unsigned short* xv    = (unsigned short*)(ws + 48 * MB);   // 16 MB
  unsigned short* xr    = (unsigned short*)(ws + 64 * MB);   // 16 MB
  float*          kbuf  = (float*)(ws + 80 * MB);            // 32 MB [T][DA]
  float*          vbuf  = (float*)(ws + 112 * MB);           // 32 MB
  float*          rbuf  = (float*)(ws + 144 * MB);           // 32 MB
  unsigned short* srwkv = (unsigned short*)(ws + 176 * MB);  // 16 MB [T][DA]
  // Chunk-state buffers reuse the xk region (dead once k/v/r are computed;
  // disjoint lifetimes within a single launch, rewritten every call).
  float* cp = (float*)(ws + 32 * MB);                        // 0.5 MB
  float* cq = (float*)(ws + 33 * MB);                        // 0.5 MB
  float* co = (float*)(ws + 34 * MB);                        // 0.5 MB

  // 1) weight convert+transpose to bf16
  const int tblocks = (2048 * 2048) / 256;
  transpose_to_bf16<<<tblocks, 256, 0, stream>>>(Wk, Wkt);
  transpose_to_bf16<<<tblocks, 256, 0, stream>>>(Wv, Wvt);
  transpose_to_bf16<<<tblocks, 256, 0, stream>>>(Wr, Wrt);
  transpose_to_bf16<<<tblocks, 256, 0, stream>>>(Wo, Wot);

  // 2) time-shift mixing -> bf16 activations
  mix_bf16<<<(T_DIM * NE_DIM) / 256, 256, 0, stream>>>(x, mk, mv, mr, xk, xv, xr);

  // 3) k / v / r projections (bf16 WMMA, f32 accum)
  dim3 ggrid(2048 / 128, 4096 / 128);
  gemm_bf16<<<ggrid, 256, 0, stream>>>(xk, Wkt, kbuf);
  gemm_bf16<<<ggrid, 256, 0, stream>>>(xv, Wvt, vbuf);
  gemm_bf16<<<ggrid, 256, 0, stream>>>(xr, Wrt, rbuf);

  // 4) chunked parallel WKV scan (serial depth 4096 -> ~192), fused sigmoid
  const int sblocks = (NCH * DA_DIM) / 256;   // 512
  wkv_chunk_state<<<sblocks, 256, 0, stream>>>(kbuf, vbuf, time_decay, cp, cq, co);
  wkv_combine<<<DA_DIM / 256, 256, 0, stream>>>(time_decay, cp, cq, co);
  wkv_chunk_apply<<<sblocks, 256, 0, stream>>>(kbuf, vbuf, rbuf, time_decay,
                                               time_first, cp, cq, co, srwkv);

  // 5) output projection -> f32 d_out
  gemm_bf16<<<ggrid, 256, 0, stream>>>(srwkv, Wot, out);
}
